// RCIPolicyNetwork_65807488909630
// MI455X (gfx1250) — matrix operand order repaired and checked by reading, not compiled, not run on hardware
//
#include <hip/hip_runtime.h>

// ---------------- problem constants ----------------
constexpr int Bn  = 512;    // batch
constexpr int OBS = 512;
constexpr int HID = 2048;
constexpr int NV  = 2048;
constexpr int Mm  = 32;
constexpr int ACT = 32;
constexpr int MAXIT = 51;
#define MU 1.5f
#define ONE_MINUS_MU (-0.5f)

typedef __bf16 bf16_t;
typedef bf16_t v16bf __attribute__((ext_vector_type(16)));
typedef bf16_t v8bf  __attribute__((ext_vector_type(8)));
typedef float  v8f   __attribute__((ext_vector_type(8)));

// ---------------- WMMA fragment helpers ----------------
// A fragment: 16x32 bf16 (MxK). lane 0-15: M=lane, e0..15 -> K=kb+e
//             lane 16-31: M=lane-16, K offset +16 (contiguous, matches B).
__device__ __forceinline__ v16bf load_a_frag(const bf16_t* __restrict__ A,
                                             size_t rowOff, int kb, int hi) {
  const v8bf* p0 = reinterpret_cast<const v8bf*>(A + rowOff + kb + hi * 16);
  const v8bf* p1 = reinterpret_cast<const v8bf*>(A + rowOff + kb + hi * 16 + 8);
  v8bf a0 = *p0;
  v8bf a1 = *p1;
  v16bf r;
#pragma unroll
  for (int i = 0; i < 8; ++i) { r[i] = a0[i]; r[i + 8] = a1[i]; }
  return r;
}

// B fragment: 32x16 bf16 (KxN), B stored TRANSPOSED row-major [N,K] so each lane
// reads 16 contiguous bf16: lane 0-15 -> col=lane, K=kb..kb+15; lane 16-31 -> col=lane-16, K=kb+16..kb+31.
__device__ __forceinline__ v16bf load_b_frag(const bf16_t* __restrict__ BT,
                                             int K, int n, int kb, int hi) {
  return *reinterpret_cast<const v16bf*>(BT + (size_t)n * K + kb + hi * 16);
}

__device__ __forceinline__ v8f wmma_bf16(v16bf a, v16bf b, v8f c) {
  return __builtin_amdgcn_wmma_f32_16x16x32_bf16(false, a, false, b, (short)0, c,
                                                 false, false);
}

// ---------------- generic WMMA GEMM: C = A(bf16,[M,K]) * WT(bf16,[N,K])^T + bias ----------------
// grid(M/16, N/512), block 256 = 8 waves; wave covers 16x64 (4 accum tiles).
template <bool RELU, bool OUTBF, bool OUTF32>
__global__ __launch_bounds__(256) void gemm_bias_act(
    const bf16_t* __restrict__ A, const bf16_t* __restrict__ WT,
    const float* __restrict__ bias, bf16_t* __restrict__ Obf,
    float* __restrict__ Of32, int M, int N, int K) {
  const int lane = threadIdx.x & 31;
  const int wave = threadIdx.x >> 5;
  const int hi = lane >> 4;
  const int lo = lane & 15;
  const int rb = blockIdx.x * 16;
  const int cb = blockIdx.y * 512 + wave * 64;

  v8f zero = {};
  v8f acc[4];
#pragma unroll
  for (int t = 0; t < 4; ++t) acc[t] = zero;

  const size_t aRow = (size_t)(rb + lo) * K;
#pragma unroll 2
  for (int kb = 0; kb < K; kb += 32) {
    // batch all fragment loads so the scheduler can clause them and overlap
    // with the WMMA chain (staggered s_wait_loadcnt instead of wait-0 per op)
    v16bf bfr[4];
#pragma unroll
    for (int t = 0; t < 4; ++t)
      bfr[t] = load_b_frag(WT, K, cb + t * 16 + lo, kb, hi);
    v16bf af = load_a_frag(A, aRow, kb, hi);
#pragma unroll
    for (int t = 0; t < 4; ++t) acc[t] = wmma_bf16(af, bfr[t], acc[t]);
  }

#pragma unroll
  for (int t = 0; t < 4; ++t) {
    const int col = cb + t * 16 + lo;
    const float bv = bias[col];
#pragma unroll
    for (int r = 0; r < 8; ++r) {
      const int row = rb + hi * 8 + r;
      float v = acc[t][r] + bv;
      if (RELU) v = fmaxf(v, 0.0f);
      const size_t idx = (size_t)row * N + col;
      if (OUTBF) Obf[idx] = (bf16_t)v;
      if (OUTF32) Of32[idx] = v;
    }
  }
}

// ---------------- main iteration: one WG = 16 waves = full 16x2048 row strip ----------------
__global__ __launch_bounds__(512) void rci_update(
    const float* __restrict__ aCur, const bf16_t* __restrict__ abf,
    const bf16_t* __restrict__ PYG, const float* __restrict__ xPt,
    const int* __restrict__ flag, float* __restrict__ aNxt,
    bf16_t* __restrict__ abfNxt) {
  constexpr int K = NV, N = NV;
  const int lane = threadIdx.x & 31;
  const int wave = threadIdx.x >> 5;  // 0..15
  const int hi = lane >> 4;
  const int lo = lane & 15;
  const int rb = blockIdx.x * 16;
  const int cb = wave * 128;

  __shared__ float part[16][16];  // [wave][row-in-strip] half-wave partial L1 sums
  __shared__ float rowsum[16];

  v8f zero = {};
  v8f acc[8];
#pragma unroll
  for (int t = 0; t < 8; ++t) acc[t] = zero;

  const size_t aRow = (size_t)(rb + lo) * K;
#pragma unroll 2
  for (int kb = 0; kb < K; kb += 32) {
    __builtin_prefetch(abf + aRow + kb + 64, 0, 1);  // global_prefetch_b8 (speculative)
    // batch all 9 fragment loads first, then the 8-WMMA chain
    v16bf bfr[8];
#pragma unroll
    for (int t = 0; t < 8; ++t)
      bfr[t] = load_b_frag(PYG, K, cb + t * 16 + lo, kb, hi);
    v16bf af = load_a_frag(abf, aRow, kb, hi);
#pragma unroll
    for (int t = 0; t < 8; ++t) acc[t] = wmma_bf16(af, bfr[t], acc[t]);
  }

  // epilogue pass 1: delta/gam; keep gam2 in acc, accumulate per-row |gam2|
  float ps[8];
#pragma unroll
  for (int r = 0; r < 8; ++r) ps[r] = 0.0f;
#pragma unroll
  for (int t = 0; t < 8; ++t) {
    const int col = cb + t * 16 + lo;
#pragma unroll
    for (int r = 0; r < 8; ++r) {
      const int row = rb + hi * 8 + r;
      const size_t idx = (size_t)row * N + col;
      const float ap = aCur[idx];
      const float d2 = ap + MU * (xPt[idx] - acc[t][r]);  // MU*delta+(1-MU)*a
      const float gm = fmaxf(d2, 0.0f);
      const float g2 = MU * gm + ONE_MINUS_MU * d2;
      acc[t][r] = g2;
      ps[r] += fabsf(g2);
    }
  }
  // deterministic reduction: shuffle across the 16 lanes sharing `hi`, then LDS tree
#pragma unroll
  for (int r = 0; r < 8; ++r) {
#pragma unroll
    for (int off = 1; off < 16; off <<= 1) ps[r] += __shfl_xor(ps[r], off, 32);
  }
  if (lo == 0) {
#pragma unroll
    for (int r = 0; r < 8; ++r) part[wave][hi * 8 + r] = ps[r];
  }
  __syncthreads();
  if (threadIdx.x < 16) {
    float s = 0.0f;
#pragma unroll
    for (int w = 0; w < 16; ++w) s += part[w][threadIdx.x];
    rowsum[threadIdx.x] = fmaxf(s, 1e-12f);
  }
  __syncthreads();

  const int act = *flag;
#pragma unroll
  for (int t = 0; t < 8; ++t) {
    const int col = cb + t * 16 + lo;
#pragma unroll
    for (int r = 0; r < 8; ++r) {
      const int row = rb + hi * 8 + r;
      const size_t idx = (size_t)row * N + col;
      const float g2 = acc[t][r];
      const float alpha = g2 / rowsum[hi * 8 + r];
      const float a2 = MU * alpha + ONE_MINUS_MU * g2;
      const float outv = act ? a2 : aCur[idx];
      aNxt[idx] = outv;
      abfNxt[idx] = (bf16_t)outv;
    }
  }
}

// ---------------- per-step active test: any ||(aY^T - x) P^T|| > TOL ----------------
__global__ __launch_bounds__(256) void rci_flag(const float* __restrict__ a,
                                                const float* __restrict__ x,
                                                const float* __restrict__ Y,
                                                const float* __restrict__ Q,
                                                int* __restrict__ flag) {
  const int lane = threadIdx.x & 31;
  const int wave = threadIdx.x >> 5;
  const int row = blockIdx.x * 8 + wave;
  const float4* ar = reinterpret_cast<const float4*>(a + (size_t)row * NV);
  const float4* yr = reinterpret_cast<const float4*>(Y + (size_t)lane * NV);
  float rj = -x[row * Mm + lane];
  for (int i = 0; i < NV / 4; ++i) {
    const float4 av = ar[i], yv = yr[i];
    rj += av.x * yv.x + av.y * yv.y + av.z * yv.z + av.w * yv.w;
  }
  __shared__ float rsh[8][32];
  rsh[wave][lane] = rj;
  __syncthreads();
  float si = 0.0f;
  for (int j = 0; j < Mm; ++j) si += Q[lane * Mm + j] * rsh[wave][j];
  float v = rj * si;  // err^2 = r Q r^T
#pragma unroll
  for (int off = 16; off > 0; off >>= 1) v += __shfl_xor(v, off, 32);
  if (lane == 0 && v > 1e-12f) atomicOr(flag, 1);
}

// ---------------- softmax over NV per row; also emits b output and bf16 copy ----------------
__global__ __launch_bounds__(256) void softmax_rows(const float* __restrict__ lg,
                                                    float* __restrict__ a0,
                                                    bf16_t* __restrict__ abf0,
                                                    float* __restrict__ outB) {
  const int lane = threadIdx.x & 31;
  const int wave = threadIdx.x >> 5;
  const int row = blockIdx.x * 8 + wave;
  const float* l = lg + (size_t)row * NV;
  float mx = -3.0e38f;
  for (int c = lane; c < NV; c += 32) mx = fmaxf(mx, l[c]);
#pragma unroll
  for (int off = 16; off > 0; off >>= 1) mx = fmaxf(mx, __shfl_xor(mx, off, 32));
  float sm = 0.0f;
  for (int c = lane; c < NV; c += 32) sm += __expf(l[c] - mx);
#pragma unroll
  for (int off = 16; off > 0; off >>= 1) sm += __shfl_xor(sm, off, 32);
  const float inv = 1.0f / sm;
  for (int c = lane; c < NV; c += 32) {
    const float e = __expf(l[c] - mx) * inv;
    const size_t idx = (size_t)row * NV + c;
    a0[idx] = e;
    abf0[idx] = (bf16_t)e;
    outB[idx] = e;
  }
}

// ---------------- small helpers ----------------
__global__ void cvt_bf16(const float* __restrict__ s, bf16_t* __restrict__ d, int n) {
  const int i = blockIdx.x * 256 + threadIdx.x;
  if (i < n) d[i] = (bf16_t)s[i];
}

__global__ void init_flags(int* __restrict__ f, int n) {
  const int i = threadIdx.x;
  if (i < n) f[i] = 0;
}

// x = state @ Gs^T   [B, M]
__global__ __launch_bounds__(256) void x_kernel(const float* __restrict__ st,
                                                const float* __restrict__ Gs,
                                                float* __restrict__ x) {
  const int lane = threadIdx.x & 31;
  const int wave = threadIdx.x >> 5;
  const int row = blockIdx.x * 8 + wave;
  const float4* ar = reinterpret_cast<const float4*>(st + (size_t)row * OBS);
  const float4* gr = reinterpret_cast<const float4*>(Gs + (size_t)lane * OBS);
  float s = 0.0f;
  for (int i = 0; i < OBS / 4; ++i) {
    const float4 av = ar[i], gv = gr[i];
    s += av.x * gv.x + av.y * gv.y + av.z * gv.z + av.w * gv.w;
  }
  x[row * Mm + lane] = s;
}

__global__ void yyt_kernel(const float* __restrict__ Y, float* __restrict__ YYt) {
  const int j = threadIdx.x, i = threadIdx.y;
  float s = 0.0f;
  for (int k = 0; k < NV; ++k) s += Y[(size_t)i * NV + k] * Y[(size_t)j * NV + k];
  YYt[i * Mm + j] = s;
}

// Gauss-Jordan 32x32 inverse (SPD, no pivoting needed), one 32x32 block
__global__ void inv32(const float* __restrict__ YYt, float* __restrict__ invM) {
  __shared__ float aug[32][65];
  const int tx = threadIdx.x, ty = threadIdx.y;
  aug[ty][tx] = YYt[ty * 32 + tx];
  aug[ty][tx + 32] = (tx == ty) ? 1.0f : 0.0f;
  __syncthreads();
  for (int p = 0; p < 32; ++p) {
    const float piv = aug[p][p];
    const float f = aug[ty][p];
    const float rp0 = aug[p][tx];
    const float rp1 = aug[p][tx + 32];
    const float c0 = aug[ty][tx];
    const float c1 = aug[ty][tx + 32];
    __syncthreads();
    if (ty == p) {
      aug[ty][tx] = rp0 / piv;
      aug[ty][tx + 32] = rp1 / piv;
    } else {
      aug[ty][tx] = c0 - (f / piv) * rp0;
      aug[ty][tx + 32] = c1 - (f / piv) * rp1;
    }
    __syncthreads();
  }
  invM[ty * 32 + tx] = aug[ty][tx + 32];
}

// P[n,m] = sum_j Y[j,n] * invM[j,m]
__global__ void p_kernel(const float* __restrict__ Y, const float* __restrict__ invM,
                         float* __restrict__ P) {
  const int id = blockIdx.x * 256 + threadIdx.x;
  if (id >= NV * Mm) return;
  const int n = id >> 5, m = id & 31;
  float s = 0.0f;
  for (int j = 0; j < Mm; ++j) s += Y[(size_t)j * NV + n] * invM[j * Mm + m];
  P[n * Mm + m] = s;
}

// Q = P^T P  [32,32]
__global__ void q_kernel(const float* __restrict__ P, float* __restrict__ Q) {
  const int j = threadIdx.x, i = threadIdx.y;
  float s = 0.0f;
  for (int n = 0; n < NV; ++n) s += P[n * Mm + i] * P[n * Mm + j];
  Q[i * Mm + j] = s;
}

// PYG = P @ Y = (PYt)^T, stored [N=2048, K=2048] row-major bf16 (WMMA B^T layout)
__global__ void pyg_kernel(const float* __restrict__ P, const float* __restrict__ Y,
                           bf16_t* __restrict__ PYG) {
  const size_t id = (size_t)blockIdx.x * 256 + threadIdx.x;
  if (id >= (size_t)NV * NV) return;
  const int n = (int)(id / NV), k = (int)(id % NV);
  float s = 0.0f;
  for (int m = 0; m < Mm; ++m) s += P[n * Mm + m] * Y[(size_t)m * NV + k];
  PYG[id] = (bf16_t)s;
}

// xPt[i,n] = sum_m x[i,m] * P[n,m]
__global__ void xpt_kernel(const float* __restrict__ x, const float* __restrict__ P,
                           float* __restrict__ xPt) {
  const size_t id = (size_t)blockIdx.x * 256 + threadIdx.x;
  if (id >= (size_t)Bn * NV) return;
  const int i = (int)(id / NV), n = (int)(id % NV);
  float s = 0.0f;
  for (int m = 0; m < Mm; ++m) s += x[i * Mm + m] * P[n * Mm + m];
  xPt[id] = s;
}

// action = a @ V^T  [B, 32]
__global__ __launch_bounds__(256) void action_kernel(const float* __restrict__ a,
                                                     const float* __restrict__ V,
                                                     float* __restrict__ out) {
  const int lane = threadIdx.x & 31;
  const int wave = threadIdx.x >> 5;
  const int row = blockIdx.x * 8 + wave;
  const float4* ar = reinterpret_cast<const float4*>(a + (size_t)row * NV);
  const float4* vr = reinterpret_cast<const float4*>(V + (size_t)lane * NV);
  float s = 0.0f;
  for (int i = 0; i < NV / 4; ++i) {
    const float4 av = ar[i], vv = vr[i];
    s += av.x * vv.x + av.y * vv.y + av.z * vv.z + av.w * vv.w;
  }
  out[row * ACT + lane] = s;
}

__global__ void sum_i(const int* __restrict__ flags, float* __restrict__ out_i) {
  if (threadIdx.x == 0) {
    int s = 0;
    for (int t = 0; t < MAXIT; ++t) s += flags[t];
    out_i[0] = (float)s;
  }
}

// ---------------- host launcher ----------------
extern "C" void kernel_launch(void* const* d_in, const int* in_sizes, int n_in,
                              void* d_out, int out_size, void* d_ws, size_t ws_size,
                              hipStream_t stream) {
  (void)in_sizes; (void)n_in; (void)out_size; (void)ws_size;
  const float* state = (const float*)d_in[0];
  const float* W1 = (const float*)d_in[1];
  const float* b1 = (const float*)d_in[2];
  const float* W2 = (const float*)d_in[3];
  const float* b2 = (const float*)d_in[4];
  const float* W3 = (const float*)d_in[5];
  const float* b3 = (const float*)d_in[6];
  const float* Y  = (const float*)d_in[7];
  const float* V  = (const float*)d_in[8];
  const float* Gs = (const float*)d_in[9];

  float* out_action = (float*)d_out;
  float* out_b = out_action + Bn * ACT;
  float* out_i = out_b + (size_t)Bn * NV;

  // workspace carve
  char* wsp = (char*)d_ws;
  size_t off = 0;
  auto carve = [&](size_t bytes) -> void* {
    void* p = wsp + off;
    off += (bytes + 255) & ~(size_t)255;
    return p;
  };
  bf16_t* statebf = (bf16_t*)carve((size_t)Bn * OBS * 2);
  bf16_t* W1bf = (bf16_t*)carve((size_t)HID * OBS * 2);
  bf16_t* W2bf = (bf16_t*)carve((size_t)HID * HID * 2);
  bf16_t* W3bf = (bf16_t*)carve((size_t)NV * HID * 2);
  bf16_t* h1bf = (bf16_t*)carve((size_t)Bn * HID * 2);
  bf16_t* h2bf = (bf16_t*)carve((size_t)Bn * HID * 2);
  float* logits = (float*)carve((size_t)Bn * NV * 4);
  float* a0 = (float*)carve((size_t)Bn * NV * 4);
  float* a1 = (float*)carve((size_t)Bn * NV * 4);
  bf16_t* abf0 = (bf16_t*)carve((size_t)Bn * NV * 2);
  bf16_t* abf1 = (bf16_t*)carve((size_t)Bn * NV * 2);
  float* xbuf = (float*)carve((size_t)Bn * Mm * 4);
  float* YYt = (float*)carve(Mm * Mm * 4);
  float* invM = (float*)carve(Mm * Mm * 4);
  float* P = (float*)carve((size_t)NV * Mm * 4);
  float* Q = (float*)carve(Mm * Mm * 4);
  bf16_t* PYG = (bf16_t*)carve((size_t)NV * NV * 2);
  float* xPt = (float*)carve((size_t)Bn * NV * 4);
  int* flags = (int*)carve(MAXIT * 4);

  // ---- precompute ----
  init_flags<<<1, 64, 0, stream>>>(flags, MAXIT);
  cvt_bf16<<<(Bn * OBS + 255) / 256, 256, 0, stream>>>(state, statebf, Bn * OBS);
  cvt_bf16<<<(HID * OBS + 255) / 256, 256, 0, stream>>>(W1, W1bf, HID * OBS);
  cvt_bf16<<<(HID * HID + 255) / 256, 256, 0, stream>>>(W2, W2bf, HID * HID);
  cvt_bf16<<<(NV * HID + 255) / 256, 256, 0, stream>>>(W3, W3bf, NV * HID);
  x_kernel<<<Bn / 8, 256, 0, stream>>>(state, Gs, xbuf);
  yyt_kernel<<<1, dim3(32, 32), 0, stream>>>(Y, YYt);
  inv32<<<1, dim3(32, 32), 0, stream>>>(YYt, invM);
  p_kernel<<<(NV * Mm + 255) / 256, 256, 0, stream>>>(Y, invM, P);
  q_kernel<<<1, dim3(32, 32), 0, stream>>>(P, Q);
  pyg_kernel<<<(int)(((size_t)NV * NV + 255) / 256), 256, 0, stream>>>(P, Y, PYG);
  xpt_kernel<<<(int)(((size_t)Bn * NV + 255) / 256), 256, 0, stream>>>(xbuf, P, xPt);

  // ---- MLP (WMMA bf16) ----
  dim3 ggrid(Bn / 16, HID / 512);
  gemm_bias_act<true, true, false><<<ggrid, 256, 0, stream>>>(
      statebf, W1bf, b1, h1bf, nullptr, Bn, HID, OBS);
  gemm_bias_act<true, true, false><<<ggrid, 256, 0, stream>>>(
      h1bf, W2bf, b2, h2bf, nullptr, Bn, HID, HID);
  gemm_bias_act<false, false, true><<<ggrid, 256, 0, stream>>>(
      h2bf, W3bf, b3, nullptr, logits, Bn, NV, HID);
  softmax_rows<<<Bn / 8, 256, 0, stream>>>(logits, a0, abf0, out_b);

  // ---- 51 fixed-point iterations ----
  float* aC = a0;
  float* aN = a1;
  bf16_t* bC = abf0;
  bf16_t* bN = abf1;
  for (int t = 0; t < MAXIT; ++t) {
    rci_flag<<<Bn / 8, 256, 0, stream>>>(aC, xbuf, Y, Q, flags + t);
    rci_update<<<Bn / 16, 512, 0, stream>>>(aC, bC, PYG, xPt, flags + t, aN, bN);
    float* tf = aC; aC = aN; aN = tf;
    bf16_t* tb = bC; bC = bN; bN = tb;
  }

  // ---- finish ----
  action_kernel<<<Bn / 8, 256, 0, stream>>>(aC, V, out_action);
  sum_i<<<1, 32, 0, stream>>>(flags, out_i);
}
